// DAML_29669634080801
// MI455X (gfx1250) — compile-verified, matrix operand order checked by reading, not run on hardware
//
#include <hip/hip_runtime.h>

// ---------------------------------------------------------------------------
// DAML pipeline for MI455X (gfx1250, wave32, WMMA).
// Heavy GEMMs (doc conv as im2col GEMM over K=3*D, cross attention over F)
// run on v_wmma_f32_16x16x32_f16 with fp32 accumulation.
// ---------------------------------------------------------------------------

typedef __attribute__((ext_vector_type(16))) _Float16 v16h;
typedef __attribute__((ext_vector_type(8)))  float    v8f;

namespace daml {
constexpr int cB  = 64;
constexpr int cL  = 500;
constexpr int cD  = 300;
constexpr int cF  = 100;
constexpr int cID = 32;

constexpr int LP      = 512;          // L padded to 16-mult
constexpr int DP      = 320;          // D padded to 32-mult (per conv tap)
constexpr int FP      = 128;          // F padded to 32-mult
constexpr int GE_ROWS = cL + 14;      // rows l = -1 .. 512  (store at l+1)
constexpr int NT      = FP / 16;      // 8 N-tiles over filters
constexpr int MT      = LP / 16;      // 32 M-tiles over positions
constexpr int KK      = DP / 32;      // 10 K-steps of 32 per conv tap
}

using namespace daml;

// ---------------------------------------------------------------------------
// Zero the halo rows of the gated-embedding buffer (rows l=-1 and l=500..512).
__global__ void k_zero_ge_pad(_Float16* ge) {
  int idx = blockIdx.x * blockDim.x + threadIdx.x;
  int total = cB * 14 * DP;
  if (idx >= total) return;
  int d = idx % DP;
  int r = (idx / DP) % 14;
  int b = idx / (DP * 14);
  int row = (r == 0) ? 0 : (cL + r);              // row 0 == l=-1 ; rows 501..513
  ge[(size_t)b * GE_ROWS * DP + (size_t)row * DP + d] = (_Float16)0.0f;
}

__global__ void k_zero_f32(float* p, int n) {
  int idx = blockIdx.x * blockDim.x + threadIdx.x;
  if (idx < n) p[idx] = 0.0f;
}

// ---------------------------------------------------------------------------
// Word-conv partial dots: t_k[b,l] = dot(e[b,l,:], wc_w[k,:])  for k=0..2.
// One block per (l,b); 3 waves, one per k; lane-strided dot + shuffle reduce.
__global__ void k_gate_dots(const float* __restrict__ emb,
                            const int*   __restrict__ doc,
                            const float* __restrict__ wcw,
                            float* __restrict__ tbuf) {
  int l = blockIdx.x, b = blockIdx.y;
  int k = threadIdx.x >> 5;
  int lane = threadIdx.x & 31;
  const float* erow = emb + (size_t)doc[b * cL + l] * cD;
  const float* wrow = wcw + k * cD;
  float s = 0.f;
  for (int d = lane; d < cD; d += 32) s += erow[d] * wrow[d];
  #pragma unroll
  for (int off = 16; off > 0; off >>= 1) s += __shfl_xor(s, off, 32);
  if (lane == 0) tbuf[((size_t)b * cL + l) * 3 + k] = s;
}

// ---------------------------------------------------------------------------
// gate[b,l] = sigmoid(t0[l-1] + t1[l] + t2[l+1] + b); write f16 gated emb.
__global__ void k_gate_apply(const float* __restrict__ emb,
                             const int*   __restrict__ doc,
                             const float* __restrict__ tbuf,
                             const float* __restrict__ wcb,
                             _Float16* __restrict__ ge) {
  int l = blockIdx.x, b = blockIdx.y;
  const float* t = tbuf + (size_t)b * cL * 3;
  float s = wcb[0] + t[l * 3 + 1];
  if (l > 0)      s += t[(l - 1) * 3 + 0];
  if (l < cL - 1) s += t[(l + 1) * 3 + 2];
  float gate = 1.0f / (1.0f + expf(-s));
  const float* erow = emb + (size_t)doc[b * cL + l] * cD;
  _Float16* grow = ge + (size_t)b * GE_ROWS * DP + (size_t)(l + 1) * DP;
  for (int d = threadIdx.x; d < DP; d += blockDim.x)
    grow[d] = (d < cD) ? (_Float16)(erow[d] * gate) : (_Float16)0.0f;
}

// ---------------------------------------------------------------------------
// Pack doc-conv weights into per-wave B-matrix fragments (f16):
// tile t = ntile*30 + seg*10 + kk ; layout [t][lane][16 contiguous halves].
__global__ void k_wpack(const float* __restrict__ dcw,
                        _Float16* __restrict__ bpack) {
  int t = blockIdx.x;                 // 0..NT*3*KK-1
  int ntile = t / (3 * KK);
  int seg   = (t / KK) % 3;
  int kk    = t % KK;
  int lane  = threadIdx.x >> 4;
  int j     = threadIdx.x & 15;
  int n     = ntile * 16 + (lane & 15);       // filter index
  int d     = kk * 32 + (lane >> 4) * 16 + j; // index within tap
  float v = 0.f;
  if (n < cF && d < cD) v = dcw[(size_t)n * 3 * cD + seg * cD + d];
  bpack[(size_t)t * 512 + lane * 16 + j] = (_Float16)v;
}

// ---------------------------------------------------------------------------
// Doc conv as WMMA GEMM: feat[b, l, f] = sum_{seg,d} ge[b,l+seg-1,d]*w + bias.
// One wave per 16x16 tile; 30 x v_wmma_f32_16x16x32_f16 per tile.
__global__ void __launch_bounds__(32)
k_docconv(const _Float16* __restrict__ ge,
          const _Float16* __restrict__ bpack,
          const float*    __restrict__ dcb,
          _Float16* __restrict__ feat) {
  int mtile = blockIdx.x & (MT - 1);
  int ntile = blockIdx.x / MT;
  int b = blockIdx.y;
  int lane = threadIdx.x;
  int mr = lane & 15;
  int kh = lane >> 4;

  v8f acc = {};
  const _Float16* geb = ge + (size_t)b * GE_ROWS * DP;
  #pragma unroll
  for (int seg = 0; seg < 3; ++seg) {
    const _Float16* arow = geb + (size_t)(mtile * 16 + mr + seg) * DP + kh * 16;
    const _Float16* bt = bpack + ((size_t)(ntile * 3 + seg) * KK) * 512 + lane * 16;
    #pragma unroll
    for (int kk = 0; kk < KK; ++kk) {
      v16h a = *(const v16h*)(arow + kk * 32);
      v16h w = *(const v16h*)(bt + (size_t)kk * 512);
      acc = __builtin_amdgcn_wmma_f32_16x16x32_f16(
          false, a, false, w, (short)0, acc, false, false);
    }
  }
  int n = ntile * 16 + mr;
  float bias = (n < cF) ? dcb[n] : 0.0f;
  _Float16* fb = feat + (size_t)b * LP * FP;
  #pragma unroll
  for (int r = 0; r < 8; ++r) {
    int row = mtile * 16 + r + (kh ? 8 : 0);
    fb[(size_t)row * FP + n] = (_Float16)(acc[r] + bias);
  }
}

// ---------------------------------------------------------------------------
// Per-position squared norms: sq[b,l] = sum_f feat[b,l,f]^2.
__global__ void k_sq(const _Float16* __restrict__ feat,
                     float* __restrict__ sq) {
  int b = blockIdx.x;
  int l = threadIdx.x;               // 0..511
  const _Float16* row = feat + ((size_t)b * LP + l) * FP;
  float s = 0.f;
  for (int f = 0; f < cF; ++f) { float v = (float)row[f]; s += v * v; }
  sq[b * LP + l] = s;
}

// ---------------------------------------------------------------------------
// Cross attention tile: c = <u_l, v_m> via WMMA (K = FP = 128, 4 steps),
// att = 1/(1+sqrt(max(u2+v2-2c,1e-12))), masked; accumulate row/col sums.
__global__ void __launch_bounds__(32)
k_att(const _Float16* __restrict__ featU,
      const _Float16* __restrict__ featI,
      const float* __restrict__ sqU,
      const float* __restrict__ sqI,
      float* __restrict__ uatt,
      float* __restrict__ iatt) {
  int ltile = blockIdx.x & (MT - 1);
  int mtile = blockIdx.x / MT;
  int b = blockIdx.y;
  int lane = threadIdx.x;
  int mr = lane & 15;
  int kh = lane >> 4;

  const _Float16* arow = featU + ((size_t)b * LP + ltile * 16 + mr) * FP + kh * 16;
  const _Float16* brow = featI + ((size_t)b * LP + mtile * 16 + mr) * FP + kh * 16;

  v8f acc = {};
  #pragma unroll
  for (int kk = 0; kk < FP / 32; ++kk) {
    v16h a = *(const v16h*)(arow + kk * 32);
    v16h bb = *(const v16h*)(brow + kk * 32);
    acc = __builtin_amdgcn_wmma_f32_16x16x32_f16(
        false, a, false, bb, (short)0, acc, false, false);
  }

  int m = mtile * 16 + mr;                 // column (item position), fixed per lane
  float v2 = sqI[b * LP + m];
  float msum = 0.f;
  #pragma unroll
  for (int r = 0; r < 8; ++r) {
    int l = ltile * 16 + r + (kh ? 8 : 0);
    float u2 = sqU[b * LP + l];
    float sqd = u2 + v2 - 2.0f * acc[r];
    float att = 1.0f / (1.0f + sqrtf(fmaxf(sqd, 1e-12f)));
    if (l >= cL || m >= cL) att = 0.0f;
    msum += att;
    // sum over the 16 lanes sharing this row (halves stay disjoint under xor<16)
    float rs = att;
    #pragma unroll
    for (int off = 1; off < 16; off <<= 1) rs += __shfl_xor(rs, off, 32);
    if (mr == 0 && l < cL) atomicAdd(&uatt[b * LP + l], rs);
  }
  if (m < cL) atomicAdd(&iatt[b * LP + m], msum);
}

// ---------------------------------------------------------------------------
// Collapsed pooling + abs-conv + mean + FC.
// mean_l(a)[f] = acb[f] + (1/498) * sum_k <acw[f,k,:], S_k>, with
// S_0 = 3T - p0 - p497 - 2p498 - 3p499
// S_1 = 3T - 2p0 - p1 - p498 - 2p499
// S_2 = 3T - 3p0 - 2p1 - p2 - p499,   T = sum_l p_l,  p_l = feat[l]*att[l].
__global__ void k_pool_fc(const _Float16* __restrict__ feat,
                          const float* __restrict__ att,
                          const float* __restrict__ acw,
                          const float* __restrict__ acb,
                          const float* __restrict__ fcw,
                          const float* __restrict__ fcb,
                          float* __restrict__ fea) {
  __shared__ float S0[cF], S1[cF], S2[cF], AM[cF];
  int b = blockIdx.x;
  int tid = threadIdx.x;

  if (tid < cF) {
    const _Float16* fb = feat + (size_t)b * LP * FP + tid;
    const float* ab = att + b * LP;
    float T = 0.f, p0 = 0.f, p1 = 0.f, p2 = 0.f, p497 = 0.f, p498 = 0.f, p499 = 0.f;
    for (int l = 0; l < cL; ++l) {
      float p = (float)fb[(size_t)l * FP] * ab[l];
      T += p;
      if (l == 0) p0 = p; else if (l == 1) p1 = p; else if (l == 2) p2 = p;
      if (l == 497) p497 = p; else if (l == 498) p498 = p; else if (l == 499) p499 = p;
    }
    S0[tid] = 3.f * T - p0 - p497 - 2.f * p498 - 3.f * p499;
    S1[tid] = 3.f * T - 2.f * p0 - p1 - p498 - 2.f * p499;
    S2[tid] = 3.f * T - 3.f * p0 - 2.f * p1 - p2 - p499;
  }
  __syncthreads();

  if (tid < cF) {
    const float* w = acw + (size_t)tid * 3 * cF;
    float s = 0.f;
    for (int g = 0; g < cF; ++g)
      s += w[g] * S0[g] + w[cF + g] * S1[g] + w[2 * cF + g] * S2[g];
    AM[tid] = acb[tid] + s * (1.0f / (float)(cL - 2));
  }
  __syncthreads();

  if (tid < cID) {
    const float* w = fcw + (size_t)tid * cF;
    float s = fcb[tid];
    for (int f = 0; f < cF; ++f) s += AM[f] * w[f];
    fea[b * cID + tid] = fmaxf(s, 0.0f);
  }
}

// ---------------------------------------------------------------------------
// Final assembly: out[0:4096] = use_fea (B,2,32), out[4096:8192] = item_fea.
__global__ void k_assemble(const float* __restrict__ feaU,
                           const float* __restrict__ feaI,
                           const int* __restrict__ uids,
                           const int* __restrict__ iids,
                           const float* __restrict__ uid_emb,
                           const float* __restrict__ iid_emb,
                           float* __restrict__ out) {
  int idx = blockIdx.x * blockDim.x + threadIdx.x;
  if (idx >= 2 * cB * 2 * cID) return;
  int half = idx >> 12;          // 0 = use_fea, 1 = item_fea
  int rem = idx & 4095;
  int b = rem >> 6;
  int s = (rem >> 5) & 1;
  int id = rem & (cID - 1);
  float v;
  if (half == 0)
    v = (s == 0) ? feaU[b * cID + id] : iid_emb[(size_t)iids[b] * cID + id];
  else
    v = (s == 0) ? feaI[b * cID + id] : uid_emb[(size_t)uids[b] * cID + id];
  out[idx] = v;
}

// ---------------------------------------------------------------------------
extern "C" void kernel_launch(void* const* d_in, const int* in_sizes, int n_in,
                              void* d_out, int out_size, void* d_ws, size_t ws_size,
                              hipStream_t stream) {
  const int*   uids     = (const int*)d_in[0];
  const int*   iids     = (const int*)d_in[1];
  const int*   udoc     = (const int*)d_in[2];
  const int*   idoc     = (const int*)d_in[3];
  const float* uemb     = (const float*)d_in[4];
  const float* iemb     = (const float*)d_in[5];
  const float* wcw      = (const float*)d_in[6];
  const float* wcb      = (const float*)d_in[7];
  const float* udcw     = (const float*)d_in[8];
  const float* udcb     = (const float*)d_in[9];
  const float* idcw     = (const float*)d_in[10];
  const float* idcb     = (const float*)d_in[11];
  const float* uacw     = (const float*)d_in[12];
  const float* uacb     = (const float*)d_in[13];
  const float* iacw     = (const float*)d_in[14];
  const float* iacb     = (const float*)d_in[15];
  const float* ufcw     = (const float*)d_in[16];
  const float* ufcb     = (const float*)d_in[17];
  const float* ifcw     = (const float*)d_in[18];
  const float* ifcb     = (const float*)d_in[19];
  const float* uid_emb  = (const float*)d_in[20];
  const float* iid_emb  = (const float*)d_in[21];
  float* out = (float*)d_out;

  // ---- workspace carve-up (256B aligned) ----
  char* base = (char*)d_ws;
  size_t off = 0;
  auto carve = [&](size_t bytes) -> char* {
    char* p = base + off;
    off = (off + bytes + 255) & ~(size_t)255;
    return p;
  };
  _Float16* geU   = (_Float16*)carve((size_t)cB * GE_ROWS * DP * 2);
  _Float16* geI   = (_Float16*)carve((size_t)cB * GE_ROWS * DP * 2);
  _Float16* bpU   = (_Float16*)carve((size_t)NT * 3 * KK * 512 * 2);
  _Float16* bpI   = (_Float16*)carve((size_t)NT * 3 * KK * 512 * 2);
  _Float16* featU = (_Float16*)carve((size_t)cB * LP * FP * 2);
  _Float16* featI = (_Float16*)carve((size_t)cB * LP * FP * 2);
  float*    tU    = (float*)carve((size_t)cB * cL * 3 * 4);
  float*    tI    = (float*)carve((size_t)cB * cL * 3 * 4);
  float*    sqU   = (float*)carve((size_t)cB * LP * 4);
  float*    sqI   = (float*)carve((size_t)cB * LP * 4);
  float*    uatt  = (float*)carve((size_t)cB * LP * 4);
  float*    iatt  = (float*)carve((size_t)cB * LP * 4);
  float*    feaU  = (float*)carve((size_t)cB * cID * 4);
  float*    feaI  = (float*)carve((size_t)cB * cID * 4);
  (void)ws_size; (void)n_in; (void)in_sizes; (void)out_size;

  // ---- stage 0: zero halos + attention accumulators ----
  {
    int n = cB * 14 * DP;
    int blocks = (n + 255) / 256;
    k_zero_ge_pad<<<blocks, 256, 0, stream>>>(geU);
    k_zero_ge_pad<<<blocks, 256, 0, stream>>>(geI);
    k_zero_f32<<<(cB * LP + 255) / 256, 256, 0, stream>>>(uatt, cB * LP);
    k_zero_f32<<<(cB * LP + 255) / 256, 256, 0, stream>>>(iatt, cB * LP);
  }

  // ---- stage 1: word gate + gated embeddings (f16) ----
  k_gate_dots<<<dim3(cL, cB), 96, 0, stream>>>(uemb, udoc, wcw, tU);
  k_gate_dots<<<dim3(cL, cB), 96, 0, stream>>>(iemb, idoc, wcw, tI);
  k_gate_apply<<<dim3(cL, cB), 128, 0, stream>>>(uemb, udoc, tU, wcb, geU);
  k_gate_apply<<<dim3(cL, cB), 128, 0, stream>>>(iemb, idoc, tI, wcb, geI);

  // ---- stage 2: pack weights, doc conv via WMMA ----
  k_wpack<<<NT * 3 * KK, 512, 0, stream>>>(udcw, bpU);
  k_wpack<<<NT * 3 * KK, 512, 0, stream>>>(idcw, bpI);
  k_docconv<<<dim3(NT * MT, cB), 32, 0, stream>>>(geU, bpU, udcb, featU);
  k_docconv<<<dim3(NT * MT, cB), 32, 0, stream>>>(geI, bpI, idcb, featI);

  // ---- stage 3: cross attention via WMMA + masked sums ----
  k_sq<<<cB, LP, 0, stream>>>(featU, sqU);
  k_sq<<<cB, LP, 0, stream>>>(featI, sqI);
  k_att<<<dim3(MT * MT, cB), 32, 0, stream>>>(featU, featI, sqU, sqI, uatt, iatt);

  // ---- stage 4: collapsed pooling + FC ----
  k_pool_fc<<<cB, 128, 0, stream>>>(featU, uatt, uacw, uacb, ufcw, ufcb, feaU);
  k_pool_fc<<<cB, 128, 0, stream>>>(featI, iatt, iacw, iacb, ifcw, ifcb, feaI);

  // ---- stage 5: assemble outputs ----
  k_assemble<<<(2 * cB * 2 * cID + 255) / 256, 256, 0, stream>>>(
      feaU, feaI, uids, iids, uid_emb, iid_emb, out);
}
